// MolmoAttention_16398185136849
// MI455X (gfx1250) — matrix operand order repaired and checked by reading, not compiled, not run on hardware
//
#include <hip/hip_runtime.h>

// ---------- constants ----------
#define B_   2
#define S_   2048
#define HID_ 2048
#define NH_  16
#define NKV_ 4
#define HD_  128
#define NREP_ 4
#define QKVN_ 3072            // HID + 2*NKV*HD
#define SCALE_ 0.08838834764831845f   // 1/sqrt(128)

// gfx1250 async global->LDS path if the toolchain exposes it
#if defined(__has_builtin)
#if __has_builtin(__builtin_amdgcn_global_load_async_to_lds_b128) && \
    __has_builtin(__builtin_amdgcn_s_wait_asynccnt)
#define HAVE_ASYNC_LDS 1
#endif
#endif

typedef __attribute__((ext_vector_type(16))) __bf16          v16bf;
typedef __attribute__((ext_vector_type(8)))  float           v8f;
typedef __attribute__((ext_vector_type(16))) unsigned short  v16u16;
typedef __attribute__((ext_vector_type(8)))  unsigned short  v8u16;
typedef __attribute__((ext_vector_type(4)))  int             v4i;

#define AS1_ __attribute__((address_space(1)))
#define AS3_ __attribute__((address_space(3)))

__device__ __forceinline__ unsigned short f2bf(float f) {
  unsigned int u = __builtin_bit_cast(unsigned int, f);
  u += 0x7FFFu + ((u >> 16) & 1u);            // round-to-nearest-even
  return (unsigned short)(u >> 16);
}

__device__ __forceinline__ v8f wmma_bf16(v16u16 a, v16u16 b, v8f c) {
  return __builtin_amdgcn_wmma_f32_16x16x32_bf16(
      false, __builtin_bit_cast(v16bf, a),
      false, __builtin_bit_cast(v16bf, b),
      (short)0, c, false, false);
}

// A-fragment gather: elements 0-7 from base[0..7], 8-15 from base[16..23]
__device__ __forceinline__ v16u16 frag_a(const unsigned short* base) {
  v8u16 lo = *(const v8u16*)(base);
  v8u16 hi = *(const v8u16*)(base + 16);
  v16u16 a;
#pragma unroll
  for (int i = 0; i < 8; ++i) { a[i] = lo[i]; a[i + 8] = hi[i]; }
  return a;
}

// 16B global -> LDS copy (async on CDNA5, sync fallback)
__device__ __forceinline__ void cp16(unsigned short* l, const unsigned short* g) {
#if HAVE_ASYNC_LDS
  __builtin_amdgcn_global_load_async_to_lds_b128(
      (AS1_ v4i*)g, (AS3_ v4i*)l, 0, 0);
#else
  *(v8u16*)l = *(const v8u16*)g;
#endif
}

// ---------- fp32 -> bf16 convert ----------
__global__ __launch_bounds__(256) void f32_to_bf16_k(
    const float* __restrict__ x, unsigned short* __restrict__ y, size_t n) {
  size_t i = (size_t)blockIdx.x * 256 + threadIdx.x;
  size_t stride = (size_t)gridDim.x * 256;
  for (; i < n; i += stride) y[i] = f2bf(x[i]);
}

// ---------- C(MxN) = A(MxK,bf16) @ W(NxK,bf16)^T ; fp32 out ----------
// one wave -> 16x64 tile; software-pipelined K loop (load k+32 while wmma k)
__global__ __launch_bounds__(256) void gemm_bf16_nt(
    const unsigned short* __restrict__ A, const unsigned short* __restrict__ W,
    float* __restrict__ C, int M, int N, int K) {
  const int lane   = threadIdx.x & 31;
  const int wave   = threadIdx.x >> 5;
  const int lane16 = lane & 15;
  const int hsel   = (lane >> 4) & 1;          // 0: lanes 0-15, 1: lanes 16-31
  const int mtile  = blockIdx.x * 8 + wave;
  const int nbase  = blockIdx.y * 64;
  if (mtile * 16 >= M) return;                  // wave-uniform

  const unsigned short* ap   = A + (size_t)(mtile * 16 + lane16) * K + hsel * 8;
  const unsigned short* wrow = W + (size_t)(nbase + lane16) * K + hsel * 8;

  v8f acc[4];
#pragma unroll
  for (int t = 0; t < 4; ++t) acc[t] = (v8f){};

  // prologue
  v16u16 a_cur = frag_a(ap);
  v16u16 b_cur[4];
#pragma unroll
  for (int t = 0; t < 4; ++t)
    b_cur[t] = frag_a(wrow + (size_t)t * 16 * K);

  for (int kk = 0; kk < K - 32; kk += 32) {
    // fetch next k-slice while current WMMAs run
    v16u16 a_nxt = frag_a(ap + kk + 32);
    v16u16 b_nxt[4];
#pragma unroll
    for (int t = 0; t < 4; ++t)
      b_nxt[t] = frag_a(wrow + (size_t)t * 16 * K + kk + 32);
    __builtin_prefetch(ap + kk + 160, 0, 0);    // global_prefetch_b8
#pragma unroll
    for (int t = 0; t < 4; ++t) acc[t] = wmma_bf16(a_cur, b_cur[t], acc[t]);
    a_cur = a_nxt;
#pragma unroll
    for (int t = 0; t < 4; ++t) b_cur[t] = b_nxt[t];
  }
#pragma unroll
  for (int t = 0; t < 4; ++t) acc[t] = wmma_bf16(a_cur, b_cur[t], acc[t]);

#pragma unroll
  for (int t = 0; t < 4; ++t)
#pragma unroll
    for (int r = 0; r < 8; ++r) {
      int m = mtile * 16 + r + hsel * 8;        // D layout: element r -> row
      int n = nbase + t * 16 + lane16;          // lane -> col
      C[(size_t)m * N + n] = acc[t][r];
    }
}

// ---------- RoPE + repack: QKV fp32 [4096,3072] -> Qb/Kb/Vt bf16 ----------
__global__ __launch_bounds__(256) void rope_pack_k(
    const float* __restrict__ QKV, const int* __restrict__ pos,
    unsigned short* __restrict__ Qb, unsigned short* __restrict__ Kb,
    unsigned short* __restrict__ Vt) {
  const int row = blockIdx.x;                 // b*S + s
  const int b = row >> 11, s = row & (S_ - 1);
  const float p = (float)pos[row];
  const float* src = QKV + (size_t)row * QKVN_;
  const float kLn = -0.14391156831212787f;    // -ln(10000)/64

  for (int idx = threadIdx.x; idx < NH_ * 64; idx += 256) {
    int h = idx >> 6, d = idx & 63;
    float x0 = src[h * HD_ + d], x1 = src[h * HD_ + d + 64];
    float fr = p * __expf(kLn * (float)d);
    float sn, cs; __sincosf(fr, &sn, &cs);
    size_t o = ((size_t)(b * NH_ + h) * S_ + s) * HD_ + d;
    Qb[o]      = f2bf(x0 * cs - x1 * sn);
    Qb[o + 64] = f2bf(x1 * cs + x0 * sn);
  }
  for (int idx = threadIdx.x; idx < NKV_ * 64; idx += 256) {
    int h = idx >> 6, d = idx & 63;
    float x0 = src[HID_ + h * HD_ + d], x1 = src[HID_ + h * HD_ + d + 64];
    float fr = p * __expf(kLn * (float)d);
    float sn, cs; __sincosf(fr, &sn, &cs);
    size_t o = ((size_t)(b * NKV_ + h) * S_ + s) * HD_ + d;
    Kb[o]      = f2bf(x0 * cs - x1 * sn);
    Kb[o + 64] = f2bf(x1 * cs + x0 * sn);
  }
  for (int idx = threadIdx.x; idx < NKV_ * HD_; idx += 256) {
    int h = idx >> 7, d = idx & (HD_ - 1);
    Vt[((size_t)(b * NKV_ + h) * HD_ + d) * S_ + s] =
        f2bf(src[HID_ + NKV_ * HD_ + idx]);
  }
}

// ---------- flash attention ----------
// block = 8 waves sharing (b,h); K/V tiles staged to LDS once per block
// (async global->LDS when available), each wave owns one 16-row q tile.
__global__ __launch_bounds__(256) void attn_fa_k(
    const unsigned short* __restrict__ Qb, const unsigned short* __restrict__ Kb,
    const unsigned short* __restrict__ Vt, unsigned short* __restrict__ AOb) {
  __shared__ __attribute__((aligned(32))) unsigned short sK[32 * HD_]; // [kv][d]
  __shared__ __attribute__((aligned(32))) unsigned short sV[HD_ * 32]; // [d][kv]

  const int tid = threadIdx.x;
  const int lane = tid & 31, wave = tid >> 5;
  const int lane16 = lane & 15, hsel = (lane >> 4) & 1;
  const int bh = blockIdx.x, b = bh >> 4, h = bh & 15, kvh = h >> 2;
  const int q0 = (blockIdx.y * 8 + wave) * 16;
  const int qrow = q0 + lane16;

  // Q^T B-fragments: B[k=d][n=q]=Q[q][d]; held for the whole kv loop
  const unsigned short* qptr = Qb + ((size_t)(b * NH_ + h) * S_ + qrow) * HD_;
  v16u16 qf[4];
#pragma unroll
  for (int ds = 0; ds < 4; ++ds)
    qf[ds] = frag_a(qptr + ds * 32 + hsel * 8);

  const unsigned short* kbp = Kb + (size_t)(b * NKV_ + kvh) * S_ * HD_;
  const unsigned short* vbp = Vt + (size_t)(b * NKV_ + kvh) * HD_ * S_;

  v8f o[8];
#pragma unroll
  for (int t = 0; t < 8; ++t) o[t] = (v8f){};
  float m_i = -1e30f, l_i = 0.f;

  const int kvend_wave = q0 + 16;                       // causal for this wave
  const int kvend_blk  = blockIdx.y * 128 + 128;        // max over block

  for (int kv0 = 0; kv0 < kvend_blk; kv0 += 32) {
    __syncthreads();                        // LDS reuse: prior compute done
    // stage K tile: rows kv0..kv0+31 are contiguous 8KB in Kb
    {
      const unsigned short* gk = kbp + (size_t)kv0 * HD_;
#pragma unroll
      for (int it = 0; it < 2; ++it) {
        int ci = tid + it * 256;            // 512 chunks of 16B
        cp16(sK + ci * 8, gk + ci * 8);
      }
      // stage V^T tile: 128 rows x 64B, row stride S
      const unsigned short* gv = vbp + kv0;
#pragma unroll
      for (int it = 0; it < 2; ++it) {
        int ci = tid + it * 256;
        int d = ci >> 2, part = ci & 3;
        cp16(sV + d * 32 + part * 8, gv + (size_t)d * S_ + part * 8);
      }
    }
#if HAVE_ASYNC_LDS
    __builtin_amdgcn_s_wait_asynccnt(0);
#endif
    __syncthreads();

    if (kv0 < kvend_wave) {                 // wave-uniform: EXEC stays all-1s
      // S^T tiles: St = K_tile(16kv x 128d) @ Q^T(128d x 16q), from LDS
      v8f st0 = (v8f){}, st1 = (v8f){};
      const unsigned short* k0p = sK + lane16 * HD_ + hsel * 8;
      const unsigned short* k1p = k0p + 16 * HD_;
#pragma unroll
      for (int ds = 0; ds < 4; ++ds) {
        st0 = wmma_bf16(frag_a(k0p + ds * 32), qf[ds], st0);
        st1 = wmma_bf16(frag_a(k1p + ds * 32), qf[ds], st1);
      }
      // causal mask + row max (row q = lane16; kv rows split by lane half)
      float p0[8], p1[8], rowm = -1e30f;
#pragma unroll
      for (int r = 0; r < 8; ++r) {
        int kp = kv0 + r + hsel * 8;
        p0[r] = (kp <= qrow)      ? st0[r] * SCALE_ : -1e9f;
        p1[r] = (kp + 16 <= qrow) ? st1[r] * SCALE_ : -1e9f;
        rowm = fmaxf(rowm, fmaxf(p0[r], p1[r]));
      }
      rowm = fmaxf(rowm, __shfl_xor(rowm, 16));
      float mnew  = fmaxf(m_i, rowm);
      float alpha = __expf(m_i - mnew);
      float rs = 0.f;
#pragma unroll
      for (int r = 0; r < 8; ++r) {
        p0[r] = __expf(p0[r] - mnew);
        p1[r] = __expf(p1[r] - mnew);
        rs += p0[r] + p1[r];
      }
      rs += __shfl_xor(rs, 16);
      l_i = l_i * alpha + rs;
      m_i = mnew;
      // P fragment: two St D-tiles ARE the bf16 A-layout of P(16q x 32kv)
      v16u16 pa;
#pragma unroll
      for (int r = 0; r < 8; ++r) { pa[r] = f2bf(p0[r]); pa[r + 8] = f2bf(p1[r]); }
      // per-O-row rescale (O element r -> row r + hsel*8)
      float av[8];
#pragma unroll
      for (int r = 0; r < 8; ++r) av[r] = __shfl(alpha, r + hsel * 8);
#pragma unroll
      for (int dt = 0; dt < 8; ++dt) {
#pragma unroll
        for (int r = 0; r < 8; ++r) o[dt][r] *= av[r];
        // V B-fragment: B[k=kv][n=d] = sV[d][kv]; table-order gather (2x16B)
        v16u16 vf = frag_a(sV + (dt * 16 + lane16) * 32 + hsel * 8);
        o[dt] = wmma_bf16(pa, vf, o[dt]);
      }
    }
  }
  // normalize + store bf16 AO[b*S+q][h*HD+d]
  float rl[8];
#pragma unroll
  for (int r = 0; r < 8; ++r) rl[r] = 1.f / __shfl(l_i, r + hsel * 8);
#pragma unroll
  for (int dt = 0; dt < 8; ++dt)
#pragma unroll
    for (int r = 0; r < 8; ++r) {
      int q = q0 + r + hsel * 8;
      AOb[(size_t)(b * S_ + q) * HID_ + h * HD_ + dt * 16 + lane16] =
          f2bf(o[dt][r] * rl[r]);
    }
}

// ---------- host ----------
extern "C" void kernel_launch(void* const* d_in, const int* in_sizes, int n_in,
                              void* d_out, int out_size, void* d_ws, size_t ws_size,
                              hipStream_t stream) {
  (void)in_sizes; (void)n_in; (void)out_size; (void)ws_size;
  const float* hs   = (const float*)d_in[0];
  const float* wqkv = (const float*)d_in[1];
  const float* wout = (const float*)d_in[2];
  /* d_in[3] attention_mask: equals causal mask, applied inline */
  const int*   pos  = (const int*)d_in[4];
  float*       out  = (float*)d_out;

  char* ws = (char*)d_ws;
  const size_t M = (size_t)B_ * S_;            // 4096
  size_t oXb  = 0;
  size_t oWq  = oXb  + M * HID_ * 2;           // X bf16
  size_t oWo  = oWq  + (size_t)QKVN_ * HID_ * 2;
  size_t oQKV = oWo  + (size_t)HID_ * HID_ * 2;
  size_t oQb  = oQKV + M * QKVN_ * 4;
  size_t oKb  = oQb  + M * HID_ * 2;
  size_t oVt  = oKb  + M * NKV_ * HD_ * 2;
  size_t oAO  = oVt  + M * NKV_ * HD_ * 2;

  unsigned short* Xb  = (unsigned short*)(ws + oXb);
  unsigned short* Wqb = (unsigned short*)(ws + oWq);
  unsigned short* Wob = (unsigned short*)(ws + oWo);
  float*          QKV = (float*)(ws + oQKV);
  unsigned short* Qb  = (unsigned short*)(ws + oQb);
  unsigned short* Kb  = (unsigned short*)(ws + oKb);
  unsigned short* Vt  = (unsigned short*)(ws + oVt);
  unsigned short* AOb = (unsigned short*)(ws + oAO);

  f32_to_bf16_k<<<2048, 256, 0, stream>>>(hs,   Xb,  M * HID_);
  f32_to_bf16_k<<<2048, 256, 0, stream>>>(wqkv, Wqb, (size_t)QKVN_ * HID_);
  f32_to_bf16_k<<<2048, 256, 0, stream>>>(wout, Wob, (size_t)HID_ * HID_);

  gemm_bf16_nt<<<dim3(32, QKVN_ / 64), 256, 0, stream>>>(
      Xb, Wqb, QKV, (int)M, QKVN_, HID_);

  rope_pack_k<<<(int)M, 256, 0, stream>>>(QKV, pos, Qb, Kb, Vt);

  attn_fa_k<<<dim3(B_ * NH_, S_ / (16 * 8)), 256, 0, stream>>>(Qb, Kb, Vt, AOb);

  gemm_bf16_nt<<<dim3(32, HID_ / 64), 256, 0, stream>>>(
      AOb, Wob, out, (int)M, HID_, HID_);
}
